// iNGPDW_89489938579893
// MI455X (gfx1250) — matrix operand order repaired and checked by reading, not compiled, not run on hardware
//
#include <hip/hip_runtime.h>
#include <hip/hip_bf16.h>

// ---------------------------------------------------------------------------
// Types for CDNA5 WMMA
// ---------------------------------------------------------------------------
typedef __attribute__((ext_vector_type(16))) _Float16     v16h;
typedef __attribute__((ext_vector_type(8)))  float        v8f;
typedef __attribute__((ext_vector_type(8)))  unsigned int v8u;

#define LVLS   10
#define TSZ    65536
#define NK     4
#define NHID   64
#define NOUT   29
#define DOFFC  (-4.0f)
#define ASCALE 1024.0f
#define AINV   (1.0f / 1024.0f)
#define WAVES  8

// floor(16 * (2^(10/9))^l) for l=0..9 (multi-res grid resolutions)
__device__ __constant__ float RESC[LVLS] = {
    16.f, 34.f, 74.f, 161.f, 348.f, 752.f, 1625.f, 3511.f, 7584.f, 16384.f};

// ---------------------------------------------------------------------------
// Kernel 1: hash-grid encode (gather-bound, table lives in L2).
// One thread per point; accumulates K=4 samples x L=10 levels x F=4 feats.
// Writes windowed mean features, layout h[b][f*10 + l], to workspace.
// ---------------------------------------------------------------------------
__global__ void __launch_bounds__(256) encode_kernel(
    const float* __restrict__ x, const float* __restrict__ cr,
    const float* __restrict__ eps, const float* __restrict__ table,
    float* __restrict__ hbuf, int B)
{
    int b = blockIdx.x * 256 + threadIdx.x;
    if (b >= B) return;

    float x0 = x[b * 3 + 0], x1 = x[b * 3 + 1], x2 = x[b * 3 + 2];
    float c  = cr[b];
    float cv0 = 4.f * x0 - 2.f, cv1 = 4.f * x1 - 2.f, cv2 = 4.f * x2 - 2.f;
    float tcr = 0.3f * c;  // TRUNC * cr

    float acc[4][LVLS];
#pragma unroll
    for (int f = 0; f < 4; ++f)
#pragma unroll
        for (int l = 0; l < LVLS; ++l) acc[f][l] = 0.f;

    const float4* tbl4 = (const float4*)table;

    for (int k = 0; k < NK; ++k) {
        float e0 = eps[(b * NK + k) * 3 + 0];
        float e1 = eps[(b * NK + k) * 3 + 1];
        float e2 = eps[(b * NK + k) * 3 + 2];
        float s0 = fminf(fmaxf(cv0 + e0 * tcr, -1.f), 1.f) * 0.25f + 0.5f;
        float s1 = fminf(fmaxf(cv1 + e1 * tcr, -1.f), 1.f) * 0.25f + 0.5f;
        float s2 = fminf(fmaxf(cv2 + e2 * tcr, -1.f), 1.f) * 0.25f + 0.5f;

#pragma unroll
        for (int l = 0; l < LVLS; ++l) {
            float r  = RESC[l];
            float sx = s0 * r, sy = s1 * r, sz = s2 * r;
            float fx = floorf(sx), fy = floorf(sy), fz = floorf(sz);
            float gx = sx - fx, gy = sy - fy, gz = sz - fz;
            unsigned cx = (unsigned)(int)fx;
            unsigned cy = (unsigned)(int)fy;
            unsigned cz = (unsigned)(int)fz;
            unsigned hx[2] = {cx, cx + 1u};
            unsigned hy[2] = {cy * 2654435761u, (cy + 1u) * 2654435761u};
            unsigned hz[2] = {cz * 805459861u, (cz + 1u) * 805459861u};
            float wx[2] = {1.f - gx, gx};
            float wy[2] = {1.f - gy, gy};
            float wz[2] = {1.f - gz, gz};
            float f0 = 0.f, f1 = 0.f, f2 = 0.f, f3 = 0.f;
#pragma unroll
            for (int ii = 0; ii < 2; ++ii)
#pragma unroll
                for (int jj = 0; jj < 2; ++jj)
#pragma unroll
                    for (int kk = 0; kk < 2; ++kk) {
                        unsigned idx = (hx[ii] ^ hy[jj] ^ hz[kk]) & (TSZ - 1u);
                        float4 fv = tbl4[(size_t)l * TSZ + idx];
                        float w = wx[ii] * wy[jj] * wz[kk];
                        f0 += w * fv.x; f1 += w * fv.y;
                        f2 += w * fv.z; f3 += w * fv.w;
                    }
            acc[0][l] += f0; acc[1][l] += f1;
            acc[2][l] += f2; acc[3][l] += f3;
        }
    }

#pragma unroll
    for (int l = 0; l < LVLS; ++l) {
        float den = 8.0f * (float)l * c;  // PLS*4*l*cr, PLS=2
        float win = erff(rsqrtf(fmaxf(den, 1e-12f)));
        float s = 0.25f * win;  // mean over K folded in
        hbuf[b * 40 + 0 * LVLS + l] = acc[0][l] * s;
        hbuf[b * 40 + 1 * LVLS + l] = acc[1][l] * s;
        hbuf[b * 40 + 2 * LVLS + l] = acc[2][l] * s;
        hbuf[b * 40 + 3 * LVLS + l] = acc[3][l] * s;
    }
}

// ---------------------------------------------------------------------------
// WMMA fragment builders (wave32, 16x16x32 f16).
// A 16x32 f16 layout: lane (m=lane&15, hi=lane>>4); dword j holds
//   K = kstep*32 + (j>>2)*16 + hi*8 + (j&3)*2  (pair K,K+1).
// B 32x16 f16 layout: lane n=lane&15, hi=lane>>4; dword j holds
//   K = kstep*32 + (j>>2)*16 + hi*8 + (j&3)*2 (pair K,K+1), column n.
// Weights stored in LDS row-major [n][k] so B pairs are k-contiguous dwords;
// the 8 ds_load_b32 merge into ds_load_b128 pairs (verified in round-1 asm).
// ---------------------------------------------------------------------------
__device__ __forceinline__ v16h load_a_frag(const unsigned int* tile_u,
                                            int m, int hi, int kstep) {
    v8u u;
#pragma unroll
    for (int j = 0; j < 8; ++j) {
        int kd = kstep * 16 + (j >> 2) * 8 + hi * 4 + (j & 3);  // dword index
        u[j] = tile_u[m * 32 + kd];
    }
    return __builtin_bit_cast(v16h, u);
}

__device__ __forceinline__ v16h load_b_frag(const unsigned int* w_u,
                                            int n, int hi, int kstep) {
    v8u u;
#pragma unroll
    for (int j = 0; j < 8; ++j) {
        int kd = kstep * 16 + (j >> 2) * 8 + hi * 4 + (j & 3);  // dword index
        u[j] = w_u[n * 32 + kd];
    }
    return __builtin_bit_cast(v16h, u);
}

// Branchless SELU: hardware v_exp_f32 path (no exec-mask churn between WMMAs).
__device__ __forceinline__ float selu_f(float v) {
    const float a = 1.6732632423543772f, s = 1.0507009873554805f;
    float e   = __expf(v);            // v_mul (log2e) + v_exp_f32
    float neg = s * a * (e - 1.f);
    float pos = s * v;
    return v > 0.f ? pos : neg;       // v_cndmask
}

// ---------------------------------------------------------------------------
// Kernel 2: MLP 40->64->64->29 via v_wmma_f32_16x16x32_f16, 16 points/wave,
// plus head epilogue (exp / tanh / SH split).
// ---------------------------------------------------------------------------
__global__ void __launch_bounds__(256) mlp_kernel(
    const float* __restrict__ hbuf,
    const float* __restrict__ W1, const float* __restrict__ b1,
    const float* __restrict__ W2, const float* __restrict__ b2,
    const float* __restrict__ W3, const float* __restrict__ b3,
    float* __restrict__ out, int B)
{
    __shared__ _Float16 w1s[64][64];        // [n][k], k 40..63 zero
    __shared__ _Float16 w2s[64][64];        // [n][k]
    __shared__ _Float16 w3s[32][64];        // [n][k], n 29..31 zero
    __shared__ float b1s[64], b2s[64], b3s[32];
    __shared__ _Float16 tA[WAVES][16][64];  // per-wave activation tile (f16)
    __shared__ float    oT[WAVES][16][32];  // per-wave raw MLP output tile

    int tid = threadIdx.x;
    for (int i = tid; i < 64 * 64; i += 256) {
        int n = i >> 6, k = i & 63;
        w1s[n][k] = (_Float16)((k < 40) ? W1[n * 40 + k] : 0.f);
        w2s[n][k] = (_Float16)W2[n * 64 + k];
    }
    for (int i = tid; i < 32 * 64; i += 256) {
        int n = i >> 6, k = i & 63;
        w3s[n][k] = (_Float16)((n < NOUT) ? W3[n * 64 + k] : 0.f);
    }
    if (tid < 64) { b1s[tid] = b1[tid]; b2s[tid] = b2[tid]; }
    if (tid < 32) b3s[tid] = (tid < NOUT) ? b3[tid] : 0.f;
    __syncthreads();

    int wave = tid >> 5, lane = tid & 31;
    int m16 = lane & 15, hi = lane >> 4;
    const unsigned int* tAu = (const unsigned int*)&tA[wave][0][0];
    const unsigned int* w1u = (const unsigned int*)&w1s[0][0];
    const unsigned int* w2u = (const unsigned int*)&w2s[0][0];
    const unsigned int* w3u = (const unsigned int*)&w3s[0][0];

    int nTiles = (B + 15) >> 4;
    for (int t = blockIdx.x * WAVES + wave; t < nTiles; t += gridDim.x * WAVES) {
        // ---- stage h tile -> f16 (scaled), pad K 40..63 with zero ----
        for (int i = lane; i < 16 * 64; i += 32) {
            int m = i >> 6, k = i & 63;
            int row = t * 16 + m;
            float v = (k < 40 && row < B) ? hbuf[row * 40 + k] * ASCALE : 0.f;
            tA[wave][m][k] = (_Float16)v;
        }
        __builtin_amdgcn_wave_barrier();

        // ---- layer 1: (16x40) x (40x64) ----
        {
            v16h a0 = load_a_frag(tAu, m16, hi, 0);
            v16h a1 = load_a_frag(tAu, m16, hi, 1);
#pragma unroll
            for (int nt = 0; nt < 4; ++nt) {
                int n = nt * 16 + m16;
                v8f cacc = {};
                v16h bb0 = load_b_frag(w1u, n, hi, 0);
                v16h bb1 = load_b_frag(w1u, n, hi, 1);
                cacc = __builtin_amdgcn_wmma_f32_16x16x32_f16(
                    false, a0, false, bb0, (short)0, cacc, false, false);
                cacc = __builtin_amdgcn_wmma_f32_16x16x32_f16(
                    false, a1, false, bb1, (short)0, cacc, false, false);
                float bias = b1s[n];
#pragma unroll
                for (int j = 0; j < 8; ++j) {
                    float v = selu_f(cacc[j] * AINV + bias);
                    tA[wave][j + 8 * hi][n] = (_Float16)(v * ASCALE);
                }
            }
        }
        __builtin_amdgcn_wave_barrier();

        // ---- layer 2: (16x64) x (64x64) ----
        {
            v16h a0 = load_a_frag(tAu, m16, hi, 0);
            v16h a1 = load_a_frag(tAu, m16, hi, 1);
#pragma unroll
            for (int nt = 0; nt < 4; ++nt) {
                int n = nt * 16 + m16;
                v8f cacc = {};
                v16h bb0 = load_b_frag(w2u, n, hi, 0);
                v16h bb1 = load_b_frag(w2u, n, hi, 1);
                cacc = __builtin_amdgcn_wmma_f32_16x16x32_f16(
                    false, a0, false, bb0, (short)0, cacc, false, false);
                cacc = __builtin_amdgcn_wmma_f32_16x16x32_f16(
                    false, a1, false, bb1, (short)0, cacc, false, false);
                float bias = b2s[n];
#pragma unroll
                for (int j = 0; j < 8; ++j) {
                    float v = selu_f(cacc[j] * AINV + bias);
                    tA[wave][j + 8 * hi][n] = (_Float16)(v * ASCALE);
                }
            }
        }
        __builtin_amdgcn_wave_barrier();

        // ---- layer 3: (16x64) x (64x32) ----
        {
            v16h a0 = load_a_frag(tAu, m16, hi, 0);
            v16h a1 = load_a_frag(tAu, m16, hi, 1);
#pragma unroll
            for (int nt = 0; nt < 2; ++nt) {
                int n = nt * 16 + m16;
                v8f cacc = {};
                v16h bb0 = load_b_frag(w3u, n, hi, 0);
                v16h bb1 = load_b_frag(w3u, n, hi, 1);
                cacc = __builtin_amdgcn_wmma_f32_16x16x32_f16(
                    false, a0, false, bb0, (short)0, cacc, false, false);
                cacc = __builtin_amdgcn_wmma_f32_16x16x32_f16(
                    false, a1, false, bb1, (short)0, cacc, false, false);
                float bias = b3s[n];
#pragma unroll
                for (int j = 0; j < 8; ++j)
                    oT[wave][j + 8 * hi][n] = cacc[j] * AINV + bias;
            }
        }
        __builtin_amdgcn_wave_barrier();

        // ---- heads: lanes 0-15 density/rgb/grd, lanes 16-31 SH ----
        {
            int m = m16;
            int b = t * 16 + m;
            if (b < B) {
                const float* row = &oT[wave][m][0];
                size_t Bs = (size_t)B;
                if (hi == 0) {
                    out[b] = expf(row[0] + DOFFC);  // density
                    float r0 = row[1] + 0.5f, r1 = row[2] + 0.5f, r2 = row[3] + 0.5f;
                    float* rgb = out + Bs;
                    rgb[b * 3 + 0] = r0; rgb[b * 3 + 1] = r1; rgb[b * 3 + 2] = r2;
                    float* grd = out + 4 * Bs;
                    float rv[3] = {r0, r1, r2};
#pragma unroll
                    for (int i = 0; i < 3; ++i)
#pragma unroll
                        for (int jj = 0; jj < 3; ++jj)
                            grd[b * 9 + i * 3 + jj] = rv[i] * tanhf(row[4 + i * 3 + jj]);
                } else {
                    float* sh = out + 13 * Bs;
#pragma unroll
                    for (int u2 = 0; u2 < 16; ++u2)
                        sh[b * 16 + u2] = row[13 + u2];
                }
            }
        }
        __builtin_amdgcn_wave_barrier();
    }
}

// ---------------------------------------------------------------------------
// Launch
// ---------------------------------------------------------------------------
extern "C" void kernel_launch(void* const* d_in, const int* in_sizes, int n_in,
                              void* d_out, int out_size, void* d_ws, size_t ws_size,
                              hipStream_t stream) {
    const float* x     = (const float*)d_in[0];
    const float* cr    = (const float*)d_in[1];
    const float* eps   = (const float*)d_in[2];
    const float* table = (const float*)d_in[3];
    const float* W1    = (const float*)d_in[4];
    const float* b1    = (const float*)d_in[5];
    const float* W2    = (const float*)d_in[6];
    const float* b2    = (const float*)d_in[7];
    const float* W3    = (const float*)d_in[8];
    const float* b3    = (const float*)d_in[9];
    float* out  = (float*)d_out;
    float* hbuf = (float*)d_ws;  // B*40 floats

    int B = in_sizes[1];  // cr has B elements
    int encBlocks = (B + 255) / 256;
    encode_kernel<<<encBlocks, 256, 0, stream>>>(x, cr, eps, table, hbuf, B);

    int nTiles = (B + 15) / 16;
    int mlpBlocks = (nTiles + WAVES - 1) / WAVES;
    if (mlpBlocks > 128) mlpBlocks = 128;
    mlp_kernel<<<mlpBlocks, 256, 0, stream>>>(hbuf, W1, b1, W2, b2, W3, b3, out, B);
}
